// Attention_51659866637080
// MI455X (gfx1250) — matrix operand order repaired and checked by reading, not compiled, not run on hardware
//
#include <hip/hip_runtime.h>
#include <math.h>

#define B_    32
#define T_    4096
#define F_    512
#define H_    512
#define MBLK  128
#define PITCH 520   // halfs per LDS row: 512 + 8 pad (row stride 1040B -> 4-bank skew)

typedef __attribute__((ext_vector_type(16))) __bf16 v16bf;
typedef __attribute__((ext_vector_type(8)))  __bf16 v8bf;
typedef __attribute__((ext_vector_type(4)))  __bf16 v4bf;
typedef __attribute__((ext_vector_type(8)))  float  v8f;

#if __has_builtin(__builtin_amdgcn_tanhf)
#define TANHF(x) __builtin_amdgcn_tanhf(x)
#else
#define TANHF(x) tanhf(x)
#endif

__device__ __forceinline__ unsigned short f2bf_u16(float f) {
  unsigned u = __builtin_bit_cast(unsigned, f);
  u += 0x7FFFu + ((u >> 16) & 1u);           // round-to-nearest-even
  return (unsigned short)(u >> 16);
}
__device__ __forceinline__ __bf16 f2bf(float f) {
  unsigned short s = f2bf_u16(f);
  return __builtin_bit_cast(__bf16, s);
}

// ---------- kernel 1: W1 [F,H] fp32 -> W1T [H,F] bf16 ----------
__global__ __launch_bounds__(256) void k_w1t(const float* __restrict__ W1,
                                             unsigned short* __restrict__ W1T) {
  int idx = blockIdx.x * 256 + threadIdx.x;   // idx = h*F + f
  int h = idx >> 9;
  int f = idx & 511;
  W1T[idx] = f2bf_u16(W1[f * H_ + h]);
}

// ---------- kernel 2: w2d [B,H] = dec [B,F] @ W2 [F,H] (fp32) ----------
__global__ __launch_bounds__(256) void k_w2d(const float* __restrict__ dec,
                                             const float* __restrict__ W2,
                                             float* __restrict__ w2d) {
  int idx = blockIdx.x * 256 + threadIdx.x;   // idx = b*H + h
  int b = idx >> 9;
  int h = idx & 511;
  float s = 0.f;
  for (int f = 0; f < F_; ++f) s += dec[b * F_ + f] * W2[f * H_ + h];
  w2d[idx] = s;
}

// ---------- kernel 3: scores[b,t] = sum_n tanh((enc@W1)[t,n] + w2d[b,n]) * V[n] ----------
__global__ __launch_bounds__(256) void k_scores(const float* __restrict__ enc,
                                                const unsigned short* __restrict__ W1Tu,
                                                const float* __restrict__ w2d,
                                                const float* __restrict__ V,
                                                float* __restrict__ out) {
  extern __shared__ char smem[];
  unsigned short* As = (unsigned short*)smem;                 // [MBLK][PITCH] bf16
  float* sc = (float*)(smem + (size_t)MBLK * PITCH * 2);      // [MBLK] partial scores

  const int tid = threadIdx.x;
  const int b  = blockIdx.x >> 5;          // T_/MBLK == 32 tiles per batch
  const int t0 = (blockIdx.x & 31) * MBLK;

  if (tid < MBLK) sc[tid] = 0.f;

  // ---- stage A block: 128 rows x 512 K, fp32 -> bf16 -> LDS (coalesced float4 loads) ----
  const float4* src = (const float4*)(enc + ((size_t)b * T_ + t0) * F_);
  for (int i = tid; i < MBLK * (F_ / 4); i += 256) {
    int row = i >> 7;                      // F_/4 == 128 float4 per row
    int c4  = i & 127;
    float4 q = src[(size_t)row * (F_ / 4) + c4];
    v4bf pk;
    pk[0] = f2bf(q.x); pk[1] = f2bf(q.y); pk[2] = f2bf(q.z); pk[3] = f2bf(q.w);
    *(v4bf*)(As + row * PITCH + c4 * 4) = pk;
  }
  __syncthreads();

  const int lane  = tid & 31;
  const int wave  = tid >> 5;
  const int nlo   = lane & 15;             // column within 16-wide N tile / row M within slice
  const int nhalf = lane >> 4;             // 0: lanes 0-15, 1: lanes 16-31
  const int koffA = nhalf * 16;            // contiguous K offset per half-wave (matches B)

  // Per-lane score partials; row assignment (rs, r) is invariant across n-tiles,
  // so accumulate locally and do the cross-lane reduction exactly once at the end.
  float acc[8][8];
  #pragma unroll
  for (int rs = 0; rs < 8; ++rs)
    #pragma unroll
    for (int r = 0; r < 8; ++r) acc[rs][r] = 0.f;

  for (int nti = 0; nti < 4; ++nti) {
    const int nt = wave * 4 + nti;         // 8 waves x 4 = 32 n-tiles (H=512)
    const int n  = nt * 16 + nlo;
    const float w2 = w2d[b * H_ + n];
    const float vv = V[n];

    v8f c[8];
    #pragma unroll
    for (int rs = 0; rs < 8; ++rs) c[rs] = (v8f){0.f,0.f,0.f,0.f,0.f,0.f,0.f,0.f};

    const unsigned short* bptr = W1Tu + (size_t)n * F_ + nhalf * 16;

    #pragma unroll 1
    for (int kt = 0; kt < 16; ++kt) {      // K = 512 in steps of 32
      const int k0 = kt * 32;
      v16bf bf = *(const v16bf*)(bptr + k0);     // 16 contiguous K for column n
      #pragma unroll
      for (int rs = 0; rs < 8; ++rs) {           // 8 row-slices of 16 rows
        const unsigned short* ap = As + (rs * 16 + nlo) * PITCH + k0 + koffA;
        v8bf alo = *(const v8bf*)ap;             // K[k0+koff .. +7]
        v8bf ahi = *(const v8bf*)(ap + 8);       // K[k0+koff+8 .. +15]
        v16bf a;
        #pragma unroll
        for (int i = 0; i < 8; ++i) { a[i] = alo[i]; a[i + 8] = ahi[i]; }
        c[rs] = __builtin_amdgcn_wmma_f32_16x16x32_bf16(
            false, a, false, bf, (short)0, c[rs], false, false);
      }
    }

    // ---- elementwise epilogue only (VALU/TRANS, no cross-lane traffic here) ----
    #pragma unroll
    for (int rs = 0; rs < 8; ++rs) {
      #pragma unroll
      for (int r = 0; r < 8; ++r) {
        acc[rs][r] += TANHF(c[rs][r] + w2) * vv;
      }
    }
  }

  // ---- single cross-lane reduction over the 16 N-lanes per half-wave ----
  #pragma unroll
  for (int rs = 0; rs < 8; ++rs) {
    #pragma unroll
    for (int r = 0; r < 8; ++r) {
      float p = acc[rs][r];
      p += __shfl_xor(p, 1, 32);
      p += __shfl_xor(p, 2, 32);
      p += __shfl_xor(p, 4, 32);
      p += __shfl_xor(p, 8, 32);
      if (nlo == 0)                        // lanes 0 (rows 0-7) and 16 (rows 8-15)
        atomicAdd(&sc[rs * 16 + nhalf * 8 + r], p);
    }
  }

  __syncthreads();
  if (tid < MBLK) out[(size_t)b * T_ + t0 + tid] = sc[tid];
}

// ---------- kernel 4: in-place softmax over T per batch ----------
__global__ __launch_bounds__(256) void k_softmax(float* __restrict__ out) {
  __shared__ float red[256];
  float* row = out + (size_t)blockIdx.x * T_;
  const int tid = threadIdx.x;

  float mx = -3.402823466e38f;
  for (int t = tid; t < T_; t += 256) mx = fmaxf(mx, row[t]);
  red[tid] = mx; __syncthreads();
  for (int s = 128; s > 0; s >>= 1) {
    if (tid < s) red[tid] = fmaxf(red[tid], red[tid + s]);
    __syncthreads();
  }
  mx = red[0]; __syncthreads();

  float sum = 0.f;
  for (int t = tid; t < T_; t += 256) sum += __expf(row[t] - mx);
  red[tid] = sum; __syncthreads();
  for (int s = 128; s > 0; s >>= 1) {
    if (tid < s) red[tid] += red[tid + s];
    __syncthreads();
  }
  const float inv = 1.f / red[0];
  for (int t = tid; t < T_; t += 256) row[t] = __expf(row[t] - mx) * inv;
}

extern "C" void kernel_launch(void* const* d_in, const int* in_sizes, int n_in,
                              void* d_out, int out_size, void* d_ws, size_t ws_size,
                              hipStream_t stream) {
  (void)in_sizes; (void)n_in; (void)out_size; (void)ws_size;
  const float* enc = (const float*)d_in[0];   // [B,T,F]
  const float* dec = (const float*)d_in[1];   // [B,F]
  const float* W1  = (const float*)d_in[2];   // [F,H]
  const float* W2  = (const float*)d_in[3];   // [F,H]
  const float* V   = (const float*)d_in[4];   // [H,1]
  float* out = (float*)d_out;                 // [B,T]

  // workspace layout: [0, 512KB) W1T bf16 ; [512KB, 576KB) w2d fp32
  unsigned short* W1T = (unsigned short*)d_ws;
  float* w2d = (float*)((char*)d_ws + (size_t)F_ * H_ * 2);

  k_w1t<<<(F_ * H_) / 256, 256, 0, stream>>>(W1, W1T);
  k_w2d<<<(B_ * H_) / 256, 256, 0, stream>>>(dec, W2, w2d);

  const size_t lds = (size_t)MBLK * PITCH * 2 + (size_t)MBLK * 4;   // ~133.6 KB
  k_scores<<<B_ * (T_ / MBLK), 256, lds, stream>>>(enc, W1T, w2d, V, out);
  k_softmax<<<B_, 256, 0, stream>>>(out);
}
